// MoEDecoderLayer_34711925686933
// MI455X (gfx1250) — matrix operand order repaired and checked by reading, not compiled, not run on hardware
//
#include <hip/hip_runtime.h>

// ---------------------------------------------------------------------------
// MI455X (gfx1250, wave32) implementation of the MoE transformer decoder layer.
// All GEMM-shaped math goes through V_WMMA_F32_16X16X32_BF16 (f32 accumulate).
// ---------------------------------------------------------------------------

typedef __bf16 bf16;
typedef __attribute__((ext_vector_type(8)))  __bf16 v8bf;
typedef __attribute__((ext_vector_type(16))) __bf16 v16bf;
typedef __attribute__((ext_vector_type(8)))  float  v8f;

#define DEV __device__ __forceinline__

constexpr int kS = 1024;      // tgt sequence
constexpr int kT = 1024;      // memory sequence
constexpr int kB = 4;         // batch
constexpr int kD = 1024;      // model dim
constexpr int kH = 16;        // heads
constexpr int kDh = 64;       // head dim
constexpr int kE = 8;         // experts
constexpr int kF = 2048;      // expert hidden
constexpr int kNTOK = kS * kB;  // 4096 tokens

// ---------------------------------------------------------------------------
// WMMA helpers (wave32 layouts per CDNA5 ISA 7.12.2)
// ---------------------------------------------------------------------------
DEV v8f wmma_bf16(v16bf a, v16bf b, v8f c) {
  // (neg_a, A, neg_b, B, c_mod, C, reuse_a, reuse_b)
  return __builtin_amdgcn_wmma_f32_16x16x32_bf16(false, a, false, b, (short)0, c,
                                                 false, false);
}

// A 16x32 bf16 fragment: lane holds row m = lane&15; elements j <-> K = 2*khi+j
// (contiguous, matching B), khi = 8*(lane>=16). rowp points at the
// row's K-window base; chunks are 16B aligned.
DEV v16bf load_a_frag(const bf16* rowp, int khi) {
  v8bf lo = *(const v8bf*)(rowp + 2 * khi);
  v8bf hi = *(const v8bf*)(rowp + 2 * khi + 8);
  v16bf r;
#pragma unroll
  for (int i = 0; i < 8; ++i) { r[i] = lo[i]; r[i + 8] = hi[i]; }
  return r;
}

// B 32x16 bf16 fragment: lane holds column n = lane&15; elements j <-> K = klo+j
// (klo = 16*(lane>=16)). With B stored transposed [N,K], the 16 elements are
// contiguous: p = BT + n*K + kk + klo.
DEV v16bf load_b_frag(const bf16* p) {
  v8bf lo = *(const v8bf*)p;
  v8bf hi = *(const v8bf*)(p + 8);
  v16bf r;
#pragma unroll
  for (int i = 0; i < 8; ++i) { r[i] = lo[i]; r[i + 8] = hi[i]; }
  return r;
}

// ---------------------------------------------------------------------------
// Elementwise staging kernels
// ---------------------------------------------------------------------------
__global__ void cast_bf16_kernel(const float* __restrict__ in, bf16* __restrict__ out, int n) {
  int i = blockIdx.x * 256 + threadIdx.x;
  if (i < n) out[i] = (bf16)in[i];
}

// in: [R,C] f32 row-major  ->  out: [C,R] bf16 row-major (i.e. W^T for B-operand)
__global__ void transpose_cast_kernel(const float* __restrict__ in, bf16* __restrict__ out,
                                      int R, int C) {
  int i = blockIdx.x * 256 + threadIdx.x;
  if (i < R * C) {
    int r = i / C, c = i % C;
    out[(size_t)c * R + r] = (bf16)in[i];
  }
}

__global__ void zero_f32_kernel(float* __restrict__ p, int n) {
  int i = blockIdx.x * 256 + threadIdx.x;
  if (i < n) p[i] = 0.f;
}

// [S*B, D] token-major bf16 -> per-head layouts.
// transpose==0: out[((b*H+h)*S + s)*64 + dh] = in * scale   ([B,H,S,Dh])
// transpose==1: out[((b*H+h)*64 + dh)*S + s] = in           ([B,H,Dh,S], for V^T)
__global__ void repack_heads_kernel(const bf16* __restrict__ in, bf16* __restrict__ out,
                                    float scale, int transpose) {
  int i = blockIdx.x * 256 + threadIdx.x;
  if (i >= kNTOK * kD) return;
  int s = i >> 12;              // / (B*D)
  int rem = i & 4095;
  int b = rem >> 10;
  int col = rem & 1023;
  int h = col >> 6;
  int dh = col & 63;
  float v = (float)in[i] * scale;
  if (!transpose)
    out[(((size_t)(b * kH + h) * kS + s) << 6) + dh] = (bf16)v;
  else
    out[(((size_t)(b * kH + h) * kDh + dh) << 10) + s] = (bf16)v;
}

// ---------------------------------------------------------------------------
// WMMA GEMM: C[M,N](bf16) = act(A[M,K](bf16) @ BT[N,K]^T + bias)
// Wave tile 32x64 (2x4 subtiles), 8 waves/block. M%32==0, N%64==0, K%32==0.
// ---------------------------------------------------------------------------
template <bool RELU>
__global__ void __launch_bounds__(256)
wmma_gemm_bf16_kernel(const bf16* __restrict__ A, const bf16* __restrict__ BT,
                      const float* __restrict__ bias, bf16* __restrict__ C,
                      int M, int N, int K) {
  const int lane = threadIdx.x & 31;
  const int wave = threadIdx.x >> 5;
  const int tiles_n = N >> 6;
  const int wtile = blockIdx.x * 8 + wave;
  const int tm = wtile / tiles_n;
  const int tn = wtile % tiles_n;
  if (tm * 32 >= M) return;
  const int row0 = tm * 32, col0 = tn * 64;
  const int khi = (lane >> 4) << 3;   // 0 or 8
  const int klo = (lane >> 4) << 4;   // 0 or 16
  const int nl = lane & 15;

  v8f acc[2][4] = {};
  const bf16* a0 = A + (size_t)(row0 + nl) * K;
  const bf16* a1 = a0 + (size_t)16 * K;

  for (int kk = 0; kk < K; kk += 32) {
    v16bf af0 = load_a_frag(a0 + kk, khi);
    v16bf af1 = load_a_frag(a1 + kk, khi);
#pragma unroll
    for (int j = 0; j < 4; ++j) {
      const bf16* bp = BT + (size_t)(col0 + j * 16 + nl) * K + kk + klo;
      v16bf bfrag = load_b_frag(bp);
      acc[0][j] = wmma_bf16(af0, bfrag, acc[0][j]);
      acc[1][j] = wmma_bf16(af1, bfrag, acc[1][j]);
    }
  }
#pragma unroll
  for (int i = 0; i < 2; ++i)
#pragma unroll
    for (int j = 0; j < 4; ++j) {
      const int n = col0 + j * 16 + nl;
      const float bv = bias[n];
#pragma unroll
      for (int r = 0; r < 8; ++r) {
        const int m = row0 + i * 16 + r + khi;   // khi == 8*(lane>=16)
        float v = acc[i][j][r] + bv;
        if (RELU) v = fmaxf(v, 0.f);
        C[(size_t)m * N + n] = (bf16)v;
      }
    }
}

// MoE second GEMM: C[M,N](f32) += gate[m,e] * (A@BT^T + bias). Wave tile skips
// entirely when no row in the 32-row tile routed to this expert.
__global__ void __launch_bounds__(256)
wmma_gemm_gated_kernel(const bf16* __restrict__ A, const bf16* __restrict__ BT,
                       const float* __restrict__ bias, const float* __restrict__ gates,
                       int expert, float* __restrict__ C, int M, int N, int K) {
  const int lane = threadIdx.x & 31;
  const int wave = threadIdx.x >> 5;
  const int tiles_n = N >> 6;
  const int wtile = blockIdx.x * 8 + wave;
  const int tm = wtile / tiles_n;
  const int tn = wtile % tiles_n;
  if (tm * 32 >= M) return;
  const int row0 = tm * 32, col0 = tn * 64;

  // one lane per row of the 32-row tile: wave-uniform skip if no row is routed
  const float glane = gates[(size_t)(row0 + lane) * kE + expert];
  if (__ballot(glane != 0.f) == 0ull) return;

  const int khi = (lane >> 4) << 3;
  const int klo = (lane >> 4) << 4;
  const int nl = lane & 15;

  v8f acc[2][4] = {};
  const bf16* a0 = A + (size_t)(row0 + nl) * K;
  const bf16* a1 = a0 + (size_t)16 * K;
  for (int kk = 0; kk < K; kk += 32) {
    v16bf af0 = load_a_frag(a0 + kk, khi);
    v16bf af1 = load_a_frag(a1 + kk, khi);
#pragma unroll
    for (int j = 0; j < 4; ++j) {
      const bf16* bp = BT + (size_t)(col0 + j * 16 + nl) * K + kk + klo;
      v16bf bfrag = load_b_frag(bp);
      acc[0][j] = wmma_bf16(af0, bfrag, acc[0][j]);
      acc[1][j] = wmma_bf16(af1, bfrag, acc[1][j]);
    }
  }
#pragma unroll
  for (int i = 0; i < 2; ++i)
#pragma unroll
    for (int j = 0; j < 4; ++j) {
      const int n = col0 + j * 16 + nl;
      const float bv = bias[n];
#pragma unroll
      for (int r = 0; r < 8; ++r) {
        const int m = row0 + i * 16 + r + khi;
        const float g = gates[(size_t)m * kE + expert];
        C[(size_t)m * N + n] += g * (acc[i][j][r] + bv);
      }
    }
}

// ---------------------------------------------------------------------------
// Fused streaming-softmax attention. One wave per 16-row Q tile, 4 waves/block.
// Q:[B*H,S,64] (pre-scaled by 1/8), K:[B*H,T,64], VT:[B*H,64,T].
// Output written as bf16 into token-major [S*B, D] (col = h*64+dh).
// ---------------------------------------------------------------------------
__global__ void __launch_bounds__(128)
attention_fused_kernel(const bf16* __restrict__ Q, const bf16* __restrict__ Kh,
                       const bf16* __restrict__ VT, bf16* __restrict__ O) {
  __shared__ __align__(16) bf16 pbuf[4][16 * 32];   // per-wave P tile staging
  const int lane = threadIdx.x & 31;
  const int wave = threadIdx.x >> 5;
  const int bh = blockIdx.y;
  const int b = bh >> 4, h = bh & 15;
  const int q0 = (blockIdx.x * 4 + wave) * 16;
  const int khi = (lane >> 4) << 3;
  const int klo = (lane >> 4) << 4;
  const int nl = lane & 15;

  // Preload Q A-fragments for K-dim windows 0..31 and 32..63
  const bf16* qrow = Q + ((size_t)bh * kS + q0 + nl) * kDh;
  const v16bf qa0 = load_a_frag(qrow, khi);
  const v16bf qa1 = load_a_frag(qrow + 32, khi);

  v8f o[4] = {};
  float mrun[8], lrun[8];
#pragma unroll
  for (int r = 0; r < 8; ++r) { mrun[r] = -3.0e38f; lrun[r] = 0.f; }

  bf16* pw = &pbuf[wave][0];

  for (int t0 = 0; t0 < kT; t0 += 32) {
    // scores for columns [t0, t0+32): two 16x16 D tiles
    v8f s0 = {}, s1 = {};
    const bf16* kb0 = Kh + ((size_t)bh * kT + t0 + nl) * kDh + klo;
    const bf16* kb1 = Kh + ((size_t)bh * kT + t0 + 16 + nl) * kDh + klo;
    s0 = wmma_bf16(qa0, load_b_frag(kb0), s0);
    s0 = wmma_bf16(qa1, load_b_frag(kb0 + 32), s0);
    s1 = wmma_bf16(qa0, load_b_frag(kb1), s1);
    s1 = wmma_bf16(qa1, load_b_frag(kb1 + 32), s1);

    // online softmax per row (row stats replicated across the 16 lanes of a half;
    // xor masks <=8 keep halves independent under wave32)
#pragma unroll
    for (int r = 0; r < 8; ++r) {
      float v0 = s0[r], v1 = s1[r];
      float rm = fmaxf(v0, v1);
      rm = fmaxf(rm, __shfl_xor(rm, 1, 32));
      rm = fmaxf(rm, __shfl_xor(rm, 2, 32));
      rm = fmaxf(rm, __shfl_xor(rm, 4, 32));
      rm = fmaxf(rm, __shfl_xor(rm, 8, 32));
      const float mn = fmaxf(mrun[r], rm);
      const float corr = __expf(mrun[r] - mn);
      const float p0 = __expf(v0 - mn), p1 = __expf(v1 - mn);
      float rs = p0 + p1;
      rs += __shfl_xor(rs, 1, 32);
      rs += __shfl_xor(rs, 2, 32);
      rs += __shfl_xor(rs, 4, 32);
      rs += __shfl_xor(rs, 8, 32);
      lrun[r] = lrun[r] * corr + rs;
      mrun[r] = mn;
#pragma unroll
      for (int t = 0; t < 4; ++t) o[t][r] *= corr;
      const int mr = r + khi;                 // D-layout row
      pw[mr * 32 + nl] = (bf16)p0;
      pw[mr * 32 + 16 + nl] = (bf16)p1;
    }
    __syncthreads();

    // P (16x32) as A-fragment from LDS, then PV into the 16x64 accumulator
    const v16bf pa = load_a_frag(pw + (size_t)nl * 32, khi);
#pragma unroll
    for (int t = 0; t < 4; ++t) {
      const bf16* vb = VT + ((size_t)bh * kDh + t * 16 + nl) * kT + t0 + klo;
      o[t] = wmma_bf16(pa, load_b_frag(vb), o[t]);
    }
    __syncthreads();
  }

  // normalize + scatter to token-major bf16 [S*B, D]
#pragma unroll
  for (int t = 0; t < 4; ++t)
#pragma unroll
    for (int r = 0; r < 8; ++r) {
      const int s = q0 + r + khi;
      const float v = o[t][r] / lrun[r];
      O[((size_t)s * kB + b) * kD + h * kDh + t * 16 + nl] = (bf16)v;
    }
}

// ---------------------------------------------------------------------------
// y = LN(resid + delta) * g + b ; emit f32 (next residual / final out) and
// optionally bf16 (next GEMM operand). One block per token row, D=1024.
// ---------------------------------------------------------------------------
template <bool DELTA_BF16>
__global__ void __launch_bounds__(256)
add_layernorm_kernel(const float* __restrict__ resid, const void* __restrict__ delta,
                     const float* __restrict__ g, const float* __restrict__ bvec,
                     float* __restrict__ out_f, bf16* __restrict__ out_bf) {
  const int row = blockIdx.x;
  const float* dr_f = (const float*)delta + (size_t)row * kD;
  const bf16* dr_b = (const bf16*)delta + (size_t)row * kD;
  __shared__ float s1[8], s2[8];
  float vals[4];
  float sum = 0.f, sq = 0.f;
#pragma unroll
  for (int i = 0; i < 4; ++i) {
    const int c = threadIdx.x + i * 256;
    const float d = DELTA_BF16 ? (float)dr_b[c] : dr_f[c];
    const float v = resid[(size_t)row * kD + c] + d;
    vals[i] = v;
    sum += v;
    sq += v * v;
  }
#pragma unroll
  for (int m = 1; m < 32; m <<= 1) {
    sum += __shfl_xor(sum, m, 32);
    sq += __shfl_xor(sq, m, 32);
  }
  if ((threadIdx.x & 31) == 0) { s1[threadIdx.x >> 5] = sum; s2[threadIdx.x >> 5] = sq; }
  __syncthreads();
  sum = 0.f; sq = 0.f;
#pragma unroll
  for (int w = 0; w < 8; ++w) { sum += s1[w]; sq += s2[w]; }
  const float mean = sum * (1.f / kD);
  const float var = sq * (1.f / kD) - mean * mean;
  const float inv = rsqrtf(var + 1e-5f);
#pragma unroll
  for (int i = 0; i < 4; ++i) {
    const int c = threadIdx.x + i * 256;
    const float y = (vals[i] - mean) * inv * g[c] + bvec[c];
    out_f[(size_t)row * kD + c] = y;
    if (out_bf) out_bf[(size_t)row * kD + c] = (bf16)y;
  }
}

// ---------------------------------------------------------------------------
// Router: logits = x @ rn_w + rn_b, deterministic top-2, softmax over the two.
// One block per token.
// ---------------------------------------------------------------------------
__global__ void __launch_bounds__(256)
router_topk_kernel(const bf16* __restrict__ X, const float* __restrict__ W,
                   const float* __restrict__ bvec, float* __restrict__ gates) {
  const int row = blockIdx.x;
  float acc[kE] = {};
  for (int d = threadIdx.x; d < kD; d += 256) {
    const float x = (float)X[(size_t)row * kD + d];
#pragma unroll
    for (int e = 0; e < kE; ++e) acc[e] += x * W[d * kE + e];
  }
#pragma unroll
  for (int e = 0; e < kE; ++e)
#pragma unroll
    for (int m = 1; m < 32; m <<= 1) acc[e] += __shfl_xor(acc[e], m, 32);
  __shared__ float sl[8][kE];
  if ((threadIdx.x & 31) == 0)
#pragma unroll
    for (int e = 0; e < kE; ++e) sl[threadIdx.x >> 5][e] = acc[e];
  __syncthreads();
  if (threadIdx.x == 0) {
    float logit[kE];
#pragma unroll
    for (int e = 0; e < kE; ++e) {
      float t = bvec[e];
#pragma unroll
      for (int w = 0; w < 8; ++w) t += sl[w][e];
      logit[e] = t;
    }
    int i0 = 0;
    for (int e = 1; e < kE; ++e) if (logit[e] > logit[i0]) i0 = e;
    int i1 = (i0 == 0) ? 1 : 0;
    for (int e = 0; e < kE; ++e) if (e != i0 && logit[e] > logit[i1]) i1 = e;
    const float mx = fmaxf(logit[i0], logit[i1]);
    const float e0 = __expf(logit[i0] - mx), e1 = __expf(logit[i1] - mx);
    const float inv = 1.f / (e0 + e1);
#pragma unroll
    for (int e = 0; e < kE; ++e) gates[(size_t)row * kE + e] = 0.f;
    gates[(size_t)row * kE + i0] = e0 * inv;
    gates[(size_t)row * kE + i1] = e1 * inv;
  }
}

// ---------------------------------------------------------------------------
// Host orchestration
// ---------------------------------------------------------------------------
static inline char* bump(char*& p, size_t bytes) {
  char* r = p;
  p += (bytes + 255) & ~(size_t)255;
  return r;
}

extern "C" void kernel_launch(void* const* d_in, const int* in_sizes, int n_in,
                              void* d_out, int out_size, void* d_ws, size_t ws_size,
                              hipStream_t stream) {
  (void)in_sizes; (void)n_in; (void)out_size; (void)ws_size;

  const float* tgt = (const float*)d_in[0];
  const float* memory = (const float*)d_in[1];
  const float* sa_w[4] = {(const float*)d_in[2], (const float*)d_in[4],
                          (const float*)d_in[6], (const float*)d_in[8]};
  const float* sa_b[4] = {(const float*)d_in[3], (const float*)d_in[5],
                          (const float*)d_in[7], (const float*)d_in[9]};
  const float* ma_w[4] = {(const float*)d_in[10], (const float*)d_in[12],
                          (const float*)d_in[14], (const float*)d_in[16]};
  const float* ma_b[4] = {(const float*)d_in[11], (const float*)d_in[13],
                          (const float*)d_in[15], (const float*)d_in[17]};
  const float* rn_w = (const float*)d_in[18];
  const float* rn_b = (const float*)d_in[19];
  const float* e_w1 = (const float*)d_in[20];
  const float* e_b1 = (const float*)d_in[21];
  const float* e_w2 = (const float*)d_in[22];
  const float* e_b2 = (const float*)d_in[23];
  const float* ln_g[3] = {(const float*)d_in[24], (const float*)d_in[26], (const float*)d_in[28]};
  const float* ln_b[3] = {(const float*)d_in[25], (const float*)d_in[27], (const float*)d_in[29]};
  float* out = (float*)d_out;

  char* p = (char*)d_ws;
  const size_t tokD = (size_t)kNTOK * kD;
  bf16* tgt_bf = (bf16*)bump(p, tokD * 2);
  bf16* mem_bf = (bf16*)bump(p, tokD * 2);
  bf16* saT[4], *maT[4];
  for (int i = 0; i < 4; ++i) saT[i] = (bf16*)bump(p, (size_t)kD * kD * 2);
  for (int i = 0; i < 4; ++i) maT[i] = (bf16*)bump(p, (size_t)kD * kD * 2);
  bf16* w1T = (bf16*)bump(p, (size_t)kE * kF * kD * 2);
  bf16* w2T = (bf16*)bump(p, (size_t)kE * kD * kF * 2);
  bf16* qp = (bf16*)bump(p, tokD * 2);
  bf16* kp = (bf16*)bump(p, tokD * 2);
  bf16* vp = (bf16*)bump(p, tokD * 2);
  bf16* qh = (bf16*)bump(p, tokD * 2);
  bf16* kh = (bf16*)bump(p, tokD * 2);
  bf16* vT = (bf16*)bump(p, tokD * 2);
  bf16* attn_bf = (bf16*)bump(p, tokD * 2);
  bf16* oproj_bf = (bf16*)bump(p, tokD * 2);
  float* x1_f = (float*)bump(p, tokD * 4);
  bf16* x1_bf = (bf16*)bump(p, tokD * 2);
  float* x2_f = (float*)bump(p, tokD * 4);
  bf16* x2_bf = (bf16*)bump(p, tokD * 2);
  float* gates = (float*)bump(p, (size_t)kNTOK * kE * 4);
  bf16* h_bf = (bf16*)bump(p, (size_t)kNTOK * kF * 2);
  float* moe_out = (float*)bump(p, tokD * 4);

  auto cdiv = [](int a, int b) { return (a + b - 1) / b; };

  // ---- stage activations / weights to bf16 (weights transposed to [N,K]) ----
  cast_bf16_kernel<<<cdiv(kNTOK * kD, 256), 256, 0, stream>>>(tgt, tgt_bf, kNTOK * kD);
  cast_bf16_kernel<<<cdiv(kNTOK * kD, 256), 256, 0, stream>>>(memory, mem_bf, kNTOK * kD);
  for (int i = 0; i < 4; ++i) {
    transpose_cast_kernel<<<cdiv(kD * kD, 256), 256, 0, stream>>>(sa_w[i], saT[i], kD, kD);
    transpose_cast_kernel<<<cdiv(kD * kD, 256), 256, 0, stream>>>(ma_w[i], maT[i], kD, kD);
  }
  for (int e = 0; e < kE; ++e) {
    transpose_cast_kernel<<<cdiv(kD * kF, 256), 256, 0, stream>>>(
        e_w1 + (size_t)e * kD * kF, w1T + (size_t)e * kF * kD, kD, kF);
    transpose_cast_kernel<<<cdiv(kF * kD, 256), 256, 0, stream>>>(
        e_w2 + (size_t)e * kF * kD, w2T + (size_t)e * kD * kF, kF, kD);
  }

  auto gemm = [&](const bf16* A, const bf16* BT, const float* bias, bf16* C,
                  int M, int N, int K, bool relu) {
    const int blocks = (M / 32) * (N / 64) / 8;
    if (relu)
      wmma_gemm_bf16_kernel<true><<<blocks, 256, 0, stream>>>(A, BT, bias, C, M, N, K);
    else
      wmma_gemm_bf16_kernel<false><<<blocks, 256, 0, stream>>>(A, BT, bias, C, M, N, K);
  };

  auto run_mha = [&](const bf16* xq, const bf16* xkv, bf16* const* wT,
                     const float* const* bias) {
    gemm(xq, wT[0], bias[0], qp, kNTOK, kD, kD, false);
    gemm(xkv, wT[1], bias[1], kp, kNTOK, kD, kD, false);
    gemm(xkv, wT[2], bias[2], vp, kNTOK, kD, kD, false);
    // per-head repack; fold 1/sqrt(Dh) into Q
    repack_heads_kernel<<<cdiv(kNTOK * kD, 256), 256, 0, stream>>>(qp, qh, 0.125f, 0);
    repack_heads_kernel<<<cdiv(kNTOK * kD, 256), 256, 0, stream>>>(kp, kh, 1.0f, 0);
    repack_heads_kernel<<<cdiv(kNTOK * kD, 256), 256, 0, stream>>>(vp, vT, 1.0f, 1);
    attention_fused_kernel<<<dim3(kS / 64, kB * kH), 128, 0, stream>>>(qh, kh, vT, attn_bf);
    gemm(attn_bf, wT[3], bias[3], oproj_bf, kNTOK, kD, kD, false);
  };

  // ---- self-attention + LN1 ----
  run_mha(tgt_bf, tgt_bf, saT, sa_b);
  add_layernorm_kernel<true><<<kNTOK, 256, 0, stream>>>(tgt, oproj_bf, ln_g[0], ln_b[0],
                                                        x1_f, x1_bf);
  // ---- cross-attention + LN2 ----
  run_mha(x1_bf, mem_bf, maT, ma_b);
  add_layernorm_kernel<true><<<kNTOK, 256, 0, stream>>>(x1_f, oproj_bf, ln_g[1], ln_b[1],
                                                        x2_f, x2_bf);
  // ---- MoE ----
  router_topk_kernel<<<kNTOK, 256, 0, stream>>>(x2_bf, rn_w, rn_b, gates);
  zero_f32_kernel<<<cdiv(kNTOK * kD, 256), 256, 0, stream>>>(moe_out, kNTOK * kD);
  for (int e = 0; e < kE; ++e) {
    gemm(x2_bf, w1T + (size_t)e * kF * kD, e_b1 + (size_t)e * kF, h_bf,
         kNTOK, kF, kD, true);
    const int blocks = (kNTOK / 32) * (kD / 64) / 8;
    wmma_gemm_gated_kernel<<<blocks, 256, 0, stream>>>(
        h_bf, w2T + (size_t)e * kD * kF, e_b2 + (size_t)e * kD, gates, e, moe_out,
        kNTOK, kD, kF);
  }
  // ---- LN3 -> final output ----
  add_layernorm_kernel<false><<<kNTOK, 256, 0, stream>>>(x2_f, moe_out, ln_g[2], ln_b[2],
                                                         out, nullptr);
}
